// DSMoE_64879775973751
// MI455X (gfx1250) — compile-verified, matrix operand order checked by reading, not run on hardware
//
#include <hip/hip_runtime.h>

#define N_TOK   8192
#define N_EMBD  256
#define D_FF    1024
#define N_EXP   32
#define CAP     8192     // max tokens per expert (power of 2: slot pack uses <<13)
#define TILE_M  32

typedef __attribute__((ext_vector_type(16))) __bf16          v16bf;
typedef __attribute__((ext_vector_type(8)))  float           v8f;
typedef __attribute__((ext_vector_type(8)))  unsigned short  v8us;
typedef __attribute__((ext_vector_type(16))) unsigned short  v16us;
typedef __attribute__((ext_vector_type(4)))  unsigned short  v4us;

static __device__ __forceinline__ unsigned short f2bf(float f) {
  unsigned int u = __builtin_bit_cast(unsigned int, f);
  u += 0x7FFFu + ((u >> 16) & 1u);       // round-to-nearest-even
  return (unsigned short)(u >> 16);
}

static __device__ __forceinline__ v16us cat16(v8us lo, v8us hi) {
  return __builtin_shufflevector(lo, hi, 0,1,2,3,4,5,6,7,8,9,10,11,12,13,14,15);
}

static __device__ __forceinline__ v8f wmma_bf16(v16us a, v16us b, v8f c) {
  return __builtin_amdgcn_wmma_f32_16x16x32_bf16(
      false, __builtin_bit_cast(v16bf, a),
      false, __builtin_bit_cast(v16bf, b),
      (short)0, c, false, false);
}

// ---------------------------------------------------------------- zero init
__global__ void zero_counts(int* __restrict__ counts) {
  if (threadIdx.x < N_EXP) counts[threadIdx.x] = 0;
}

// ---------------------------------------------------- 32-entry prefix scan
__global__ void scan_kernel(const int* __restrict__ counts, int* __restrict__ offs) {
  const int lane = threadIdx.x;   // 32 threads
  int sum = 0;
  for (int i = 0; i < lane; ++i) sum += counts[i];
  offs[lane] = sum;               // exclusive prefix
}

// ------------------------------------------- fp32 [R][Cc] -> bf16 [Cc][R]
__global__ __launch_bounds__(256) void transpose_f32_to_bf16(
    const float* __restrict__ src, unsigned short* __restrict__ dst,
    int R, int Cc) {
  __shared__ float t[32][33];
  const size_t eoff = (size_t)blockIdx.z * (size_t)R * (size_t)Cc;
  const int gx = blockIdx.x * 32;   // column base (Cc dim)
  const int gy = blockIdx.y * 32;   // row base    (R dim)
  const int tx = threadIdx.x;       // 0..31
  const int ty = threadIdx.y;       // 0..7
  #pragma unroll
  for (int i = ty; i < 32; i += 8)
    t[i][tx] = src[eoff + (size_t)(gy + i) * Cc + gx + tx];
  __syncthreads();
  #pragma unroll
  for (int i = ty; i < 32; i += 8)
    dst[eoff + (size_t)(gx + i) * R + gy + tx] = f2bf(t[tx][i]);
}

// ------------------------------------------------------------------ router
// one wave (32 lanes) per token; lane = expert
__global__ __launch_bounds__(256) void router_kernel(
    const float* __restrict__ x, const float* __restrict__ Wg,
    float* __restrict__ router_out, int* __restrict__ counts,
    int* __restrict__ tokIdx, float* __restrict__ tokW,
    int* __restrict__ slotOf) {
  const int lane = threadIdx.x & 31;
  const int wave = threadIdx.x >> 5;
  const int n = blockIdx.x * 8 + wave;
  const float* xr = x + (size_t)n * N_EMBD;

  float logit = 0.0f;
  for (int c = 0; c < N_EMBD; c += 4) {
    float4 xv = *(const float4*)(xr + c);
    logit += xv.x * Wg[(c + 0) * N_EXP + lane];
    logit += xv.y * Wg[(c + 1) * N_EXP + lane];
    logit += xv.z * Wg[(c + 2) * N_EXP + lane];
    logit += xv.w * Wg[(c + 3) * N_EXP + lane];
  }
  // softmax across the 32 lanes (wave32)
  float m = logit;
  for (int off = 16; off >= 1; off >>= 1) m = fmaxf(m, __shfl_xor(m, off, 32));
  float p = expf(logit - m);
  float s = p;
  for (int off = 16; off >= 1; off >>= 1) s += __shfl_xor(s, off, 32);
  float prob = p / s;

  // top-1 (argmax, lowest index on tie)
  float v1 = prob; int i1 = lane;
  for (int off = 16; off >= 1; off >>= 1) {
    float ov = __shfl_xor(v1, off, 32); int oi = __shfl_xor(i1, off, 32);
    if (ov > v1 || (ov == v1 && oi < i1)) { v1 = ov; i1 = oi; }
  }
  // top-2
  float v2 = (lane == i1) ? -1.0f : prob; int i2 = lane;
  for (int off = 16; off >= 1; off >>= 1) {
    float ov = __shfl_xor(v2, off, 32); int oi = __shfl_xor(i2, off, 32);
    if (ov > v2 || (ov == v2 && oi < i2)) { v2 = ov; i2 = oi; }
  }
  const float norm = fmaxf(v1 + v2, 1e-6f);
  const float w1 = v1 / norm, w2 = v2 / norm;

  router_out[(size_t)n * N_EXP + lane] =
      (lane == i1) ? w1 : ((lane == i2) ? w2 : 0.0f);

  if (lane == 0) {
    int pos = atomicAdd(&counts[i1], 1);
    tokIdx[i1 * CAP + pos] = n; tokW[i1 * CAP + pos] = w1;
    slotOf[n * 2 + 0] = (i1 << 13) | pos;
  } else if (lane == 1) {
    int pos = atomicAdd(&counts[i2], 1);
    tokIdx[i2 * CAP + pos] = n; tokW[i2 * CAP + pos] = w2;
    slotOf[n * 2 + 1] = (i2 << 13) | pos;
  }
}

// ----------------------------------------------------- fused expert GEMMs
// 32-token tile per block; ff processed in 4 chunks of 256 with
// double-buffered LDS h-stage; phase-2 accumulators live across chunks.
#define SX_STRIDE 264   // 256 + 8 pad (conflict-free b128 A-loads)

__global__ __launch_bounds__(128) void moe_expert_tile_kernel(
    const float* __restrict__ x,
    const unsigned short* __restrict__ W1t,   // [E][F][C] bf16
    const unsigned short* __restrict__ W2t,   // [E][C][F] bf16
    const int* __restrict__ counts,
    const int* __restrict__ offs,
    const int* __restrict__ tokIdx,
    float* __restrict__ yslot) {              // packed [offs[e]+pos][256]
  const int e    = blockIdx.y;
  const int cnt  = counts[e];
  const int base = blockIdx.x * TILE_M;
  if (base >= cnt) return;
  const int gbase = offs[e] + base;

  __shared__ __align__(16) unsigned short sX[TILE_M][SX_STRIDE];
  __shared__ __align__(16) unsigned short sH[2][TILE_M][SX_STRIDE];

  const int tid  = threadIdx.x;
  const int lane = tid & 31;
  const int wave = tid >> 5;
  const int hf   = lane >> 4;       // half-wave 0/1
  const int ln   = lane & 15;
  const int koff = hf * 8;          // CDNA5 16-bit A/B K split per half-wave
  const int cb   = wave * 64;       // output channels owned by this wave

  // ---- gather 32 token rows (fp32 -> bf16) into LDS
  {
    const int r  = tid >> 2;        // 0..31
    const int c0 = (tid & 3) * 64;  // 64 columns per thread
    const int s  = base + r;
    if (s < cnt) {
      const int tok = tokIdx[e * CAP + s];
      const float4* xr = (const float4*)(x + (size_t)tok * N_EMBD + c0);
      #pragma unroll
      for (int i = 0; i < 16; ++i) {
        float4 v = xr[i];
        v4us bv;
        bv[0] = f2bf(v.x); bv[1] = f2bf(v.y); bv[2] = f2bf(v.z); bv[3] = f2bf(v.w);
        *(v4us*)&sX[r][c0 + i * 4] = bv;
      }
    } else {
      v4us z = (v4us)0;
      #pragma unroll
      for (int i = 0; i < 16; ++i) *(v4us*)&sX[r][c0 + i * 4] = z;
    }
  }
  __syncthreads();

  v8f oacc[2][4];
  #pragma unroll
  for (int t = 0; t < 2; ++t)
    #pragma unroll
    for (int j = 0; j < 4; ++j) oacc[t][j] = (v8f)0.0f;

  for (int fo = 0; fo < 4; ++fo) {
    const int buf = fo & 1;

    // ---- phase 1: sH[buf][32][256] = relu(X @ W1[e][:, fo*256 .. +255])^2
    #pragma unroll
    for (int fci = 0; fci < 2; ++fci) {
      const int fl = (wave * 2 + fci) * 32;   // within chunk
      const int fg = fo * 256 + fl;           // global ff column base
      v8f a00 = (v8f)0.0f, a01 = (v8f)0.0f, a10 = (v8f)0.0f, a11 = (v8f)0.0f;
      const unsigned short* b0row = W1t + ((size_t)e * D_FF + fg + ln) * N_EMBD;
      const unsigned short* b1row = b0row + (size_t)16 * N_EMBD;
      #pragma unroll
      for (int kc = 0; kc < 8; ++kc) {
        const int kb = kc * 32;
        v16us A0 = cat16(*(const v8us*)&sX[ln][kb + koff],
                         *(const v8us*)&sX[ln][kb + 16 + koff]);
        v16us A1 = cat16(*(const v8us*)&sX[16 + ln][kb + koff],
                         *(const v8us*)&sX[16 + ln][kb + 16 + koff]);
        v16us B0 = cat16(*(const v8us*)(b0row + kb + koff),
                         *(const v8us*)(b0row + kb + 16 + koff));
        v16us B1 = cat16(*(const v8us*)(b1row + kb + koff),
                         *(const v8us*)(b1row + kb + 16 + koff));
        a00 = wmma_bf16(A0, B0, a00);
        a10 = wmma_bf16(A1, B0, a10);
        a01 = wmma_bf16(A0, B1, a01);
        a11 = wmma_bf16(A1, B1, a11);
      }
      const int m0 = hf * 8;
      #pragma unroll
      for (int v = 0; v < 8; ++v) {
        float h;
        h = a00[v]; sH[buf][m0 + v][fl + ln]           = f2bf(h > 0.f ? h * h : 0.f);
        h = a01[v]; sH[buf][m0 + v][fl + 16 + ln]      = f2bf(h > 0.f ? h * h : 0.f);
        h = a10[v]; sH[buf][16 + m0 + v][fl + ln]      = f2bf(h > 0.f ? h * h : 0.f);
        h = a11[v]; sH[buf][16 + m0 + v][fl + 16 + ln] = f2bf(h > 0.f ? h * h : 0.f);
      }
    }
    __syncthreads();

    // ---- phase 2: oacc += sH[buf] @ W2[e][fo*256 .. +255][cb .. cb+63]
    #pragma unroll
    for (int kc = 0; kc < 8; ++kc) {
      const int kb = kc * 32;
      const int kg = fo * 256 + kb;
      v16us A0 = cat16(*(const v8us*)&sH[buf][ln][kb + koff],
                       *(const v8us*)&sH[buf][ln][kb + 16 + koff]);
      v16us A1 = cat16(*(const v8us*)&sH[buf][16 + ln][kb + koff],
                       *(const v8us*)&sH[buf][16 + ln][kb + 16 + koff]);
      #pragma unroll
      for (int j = 0; j < 4; ++j) {
        const unsigned short* brow =
            W2t + ((size_t)e * N_EMBD + cb + j * 16 + ln) * D_FF;
        v16us B = cat16(*(const v8us*)(brow + kg + koff),
                        *(const v8us*)(brow + kg + 16 + koff));
        oacc[0][j] = wmma_bf16(A0, B, oacc[0][j]);
        oacc[1][j] = wmma_bf16(A1, B, oacc[1][j]);
      }
    }
  }

  // ---- epilogue: plain stores into packed slot outputs (no atomics)
  #pragma unroll
  for (int t = 0; t < 2; ++t) {
    #pragma unroll
    for (int v = 0; v < 8; ++v) {
      const int m = t * 16 + hf * 8 + v;     // D row within 32-token tile
      if (base + m < cnt) {
        float* yr = yslot + (size_t)(gbase + m) * N_EMBD + cb;
        #pragma unroll
        for (int j = 0; j < 4; ++j) yr[j * 16 + ln] = oacc[t][j][v];
      }
    }
  }
}

// --------------------------------------------- deterministic top-2 combine
__global__ __launch_bounds__(256) void combine_kernel(
    const float* __restrict__ yslot, const int* __restrict__ slotOf,
    const int* __restrict__ offs, const float* __restrict__ tokW,
    float* __restrict__ out) {
  const int idx = blockIdx.x * 256 + threadIdx.x;   // n*256 + c
  const int n = idx >> 8;
  const int c = idx & (N_EMBD - 1);
  const int s1 = slotOf[n * 2 + 0];                 // (e<<13)|pos == tokW index
  const int s2 = slotOf[n * 2 + 1];
  const int g1 = offs[s1 >> 13] + (s1 & (CAP - 1));
  const int g2 = offs[s2 >> 13] + (s2 & (CAP - 1));
  out[idx] = tokW[s1] * yslot[(size_t)g1 * N_EMBD + c] +
             tokW[s2] * yslot[(size_t)g2 * N_EMBD + c];
}

// ------------------------------------------------------------------ launch
extern "C" void kernel_launch(void* const* d_in, const int* in_sizes, int n_in,
                              void* d_out, int out_size, void* d_ws, size_t ws_size,
                              hipStream_t stream) {
  (void)in_sizes; (void)n_in; (void)out_size; (void)ws_size;
  const float* x  = (const float*)d_in[0];   // [4,2048,256]
  const float* Wg = (const float*)d_in[1];   // [256,32]
  const float* W1 = (const float*)d_in[2];   // [32,256,1024]
  const float* W2 = (const float*)d_in[3];   // [32,1024,256]

  float* out0 = (float*)d_out;                       // [8192][256]
  float* rout = out0 + (size_t)N_TOK * N_EMBD;       // [8192][32]

  char* ws = (char*)d_ws;
  size_t off = 0;
  int*   counts = (int*)(ws + off);            off += 256;
  int*   offs   = (int*)(ws + off);            off += 256;
  int*   tokIdx = (int*)(ws + off);            off += (size_t)N_EXP * CAP * 4;
  float* tokW   = (float*)(ws + off);          off += (size_t)N_EXP * CAP * 4;
  int*   slotOf = (int*)(ws + off);            off += (size_t)N_TOK * 2 * 4;
  unsigned short* W1t = (unsigned short*)(ws + off);
  off += (size_t)N_EXP * D_FF * N_EMBD * 2;
  unsigned short* W2t = (unsigned short*)(ws + off);
  off += (size_t)N_EXP * N_EMBD * D_FF * 2;
  float* yslot = (float*)(ws + off);           // [2*N_TOK][256] packed

  zero_counts<<<1, 32, 0, stream>>>(counts);

  // W1 [E][256][1024] -> W1t [E][1024][256]
  transpose_f32_to_bf16<<<dim3(D_FF / 32, N_EMBD / 32, N_EXP), dim3(32, 8), 0, stream>>>(
      W1, W1t, N_EMBD, D_FF);
  // W2 [E][1024][256] -> W2t [E][256][1024]
  transpose_f32_to_bf16<<<dim3(N_EMBD / 32, D_FF / 32, N_EXP), dim3(32, 8), 0, stream>>>(
      W2, W2t, D_FF, N_EMBD);

  router_kernel<<<N_TOK / 8, 256, 0, stream>>>(x, Wg, rout, counts, tokIdx, tokW, slotOf);

  scan_kernel<<<1, 32, 0, stream>>>(counts, offs);

  moe_expert_tile_kernel<<<dim3(CAP / TILE_M, N_EXP), 128, 0, stream>>>(
      x, W1t, W2t, counts, offs, tokIdx, yslot);

  combine_kernel<<<(N_TOK * N_EMBD) / 256, 256, 0, stream>>>(
      yslot, slotOf, offs, tokW, out0);
}
